// CartesianDecomposedAttention_876173328862
// MI455X (gfx1250) — matrix operand order repaired and checked by reading, not compiled, not run on hardware
//
#include <hip/hip_runtime.h>
#include <math.h>

// ---------------------------------------------------------------------------
// Complex attention (RoPE rotors, complex scores, real softmax * unit phase)
// for MI455X / gfx1250.  All matmuls run on v_wmma_f32_16x16x32_bf16 using a
// bf16x3 split (value = hi + lo, compute hi*hi + hi*lo + lo*hi with fp32
// accumulation) for near-fp32 accuracy at bf16 WMMA throughput.
//
// Roofline: ~103 GFLOP vs ~64MB unique data (L2-resident on the 192MB L2)
// -> compute bound.  All shared operands (x, weights, Q, K, V, attn ctx) are
// pre-split ONCE into bf16 hi/lo planes so the hot loops are pure b128 loads
// + WMMAs.  Projection kernels use 2x2 register blocking (32x32 per wave):
// 16 fragment loads feed 48 WMMAs per K-chunk, keeping the tensor pipes fed
// from L0/L2 instead of being load-limited.
// ---------------------------------------------------------------------------

#define D_MODEL 1024
#define NHEADS  16
#define DH      64
#define BATCH   2
#define SEQ     1024
#define MROWS   (BATCH * SEQ)   // 2048

typedef __attribute__((ext_vector_type(16))) __bf16 v16bf;
typedef __attribute__((ext_vector_type(8)))  __bf16 v8bf;
typedef __attribute__((ext_vector_type(8)))  float  v8f;

// Split-precision bf16 fragment: value ~= hi + lo
struct Frag { v16bf hi; v16bf lo; };

static __device__ __forceinline__ v16bf cat8(v8bf a, v8bf b) {
  return __builtin_shufflevector(a, b, 0, 1, 2, 3, 4, 5, 6, 7,
                                       8, 9, 10, 11, 12, 13, 14, 15);
}

static __device__ __forceinline__ void frag_set(Frag& f, int e, float x) {
  __bf16 h = (__bf16)x;
  f.hi[e] = h;
  f.lo[e] = (__bf16)(x - (float)h);
}

static __device__ __forceinline__ void store_split(__bf16* __restrict__ hi,
                                                   __bf16* __restrict__ lo,
                                                   size_t idx, float v) {
  __bf16 h = (__bf16)v;
  hi[idx] = h;
  lo[idx] = (__bf16)(v - (float)h);
}

static __device__ __forceinline__ v8f vzero8() {
  v8f z;
#pragma unroll
  for (int i = 0; i < 8; ++i) z[i] = 0.0f;
  return z;
}

// bf16x3: c += A_hi*B_hi + A_hi*B_lo + A_lo*B_hi   (fp32 accumulate)
static __device__ __forceinline__ v8f wmma_bf16x3(const Frag& a, const Frag& b, v8f c) {
  c = __builtin_amdgcn_wmma_f32_16x16x32_bf16(false, a.hi, false, b.hi, (short)0, c, false, false);
  c = __builtin_amdgcn_wmma_f32_16x16x32_bf16(false, a.hi, false, b.lo, (short)0, c, false, false);
  c = __builtin_amdgcn_wmma_f32_16x16x32_bf16(false, a.lo, false, b.hi, (short)0, c, false, false);
  return c;
}

// A fragment (M=16 x K=32) from pre-split bf16 planes, row-major [rows, ld].
// ISA layout: lane m = lane%16; k-runs [8*half, +8) and [16+8*half, +8).
static __device__ __forceinline__ Frag load_a_frag(const __bf16* __restrict__ hi,
                                                   const __bf16* __restrict__ lo,
                                                   int ld, int m0, int k0) {
  const int lane = threadIdx.x & 31;
  const int hlf  = lane >> 4;
  const size_t off = (size_t)(m0 + (lane & 15)) * ld + k0;
  const __bf16* ph = hi + off;
  const __bf16* pl = lo + off;
  Frag f;
  f.hi = cat8(*(const v8bf*)(ph + 8 * hlf), *(const v8bf*)(ph + 16 + 8 * hlf));
  f.lo = cat8(*(const v8bf*)(pl + 8 * hlf), *(const v8bf*)(pl + 16 + 8 * hlf));
  return f;
}

// B fragment (K=32 x N=16) from planes stored row-major [N, K] (B[k][n]=src[n][k]).
// ISA layout: lane n = lane%16; k-runs [8*half, +8) and [16+8*half, +8).
static __device__ __forceinline__ Frag load_b_frag(const __bf16* __restrict__ hi,
                                                   const __bf16* __restrict__ lo,
                                                   int ld, int n0, int k0) {
  const int lane = threadIdx.x & 31;
  const int hlf  = lane >> 4;
  const size_t off = (size_t)(n0 + (lane & 15)) * ld + k0;
  const __bf16* ph = hi + off;
  const __bf16* pl = lo + off;
  Frag f;
  f.hi = cat8(*(const v8bf*)(ph + 8 * hlf), *(const v8bf*)(ph + 16 + 8 * hlf));
  f.lo = cat8(*(const v8bf*)(pl + 8 * hlf), *(const v8bf*)(pl + 16 + 8 * hlf));
  return f;
}

// ---------------------------------------------------------------------------
// Kernel 0: fp32 -> bf16 (hi, lo) split, 4 elements/thread, memory bound.
// ---------------------------------------------------------------------------
__global__ __launch_bounds__(256) void split_bf16_kernel(
    const float* __restrict__ src, __bf16* __restrict__ hi,
    __bf16* __restrict__ lo, int n4) {
  const int i = blockIdx.x * 256 + threadIdx.x;
  if (i >= n4) return;
  const float4 v = ((const float4*)src)[i];
  float x[4] = {v.x, v.y, v.z, v.w};
  typedef __attribute__((ext_vector_type(4))) __bf16 v4bf;
  v4bf h, l;
#pragma unroll
  for (int e = 0; e < 4; ++e) {
    __bf16 hh = (__bf16)x[e];
    h[e] = hh;
    l[e] = (__bf16)(x[e] - (float)hh);
  }
  ((v4bf*)hi)[i] = h;
  ((v4bf*)lo)[i] = l;
}

// ---------------------------------------------------------------------------
// Kernel 1: fused complex QKV projection + RoPE rotors, 2x2 register blocked.
//   y = x @ w^T (complex): yre = xr*wr - xi*wi ; yim = xr*wi + xi*wr
// Each wave computes a 32x32 output patch (4 complex 16x16 tiles).
// Q/K written rotated as bf16 hi/lo planes [B,H,S,Dh].  V written transposed
// [B,H,Dh,S] so the attention kernel's B-fragments are contiguous loads.
// ---------------------------------------------------------------------------
__global__ __launch_bounds__(256) void qkv_rope_kernel(
    const __bf16* __restrict__ xr_hi, const __bf16* __restrict__ xr_lo,
    const __bf16* __restrict__ xi_hi, const __bf16* __restrict__ xi_lo,
    const __bf16* __restrict__ wqr_hi, const __bf16* __restrict__ wqr_lo,
    const __bf16* __restrict__ wqi_hi, const __bf16* __restrict__ wqi_lo,
    const __bf16* __restrict__ wkr_hi, const __bf16* __restrict__ wkr_lo,
    const __bf16* __restrict__ wki_hi, const __bf16* __restrict__ wki_lo,
    const __bf16* __restrict__ wvr_hi, const __bf16* __restrict__ wvr_lo,
    const __bf16* __restrict__ wvi_hi, const __bf16* __restrict__ wvi_lo,
    __bf16* __restrict__ qr_hi, __bf16* __restrict__ qr_lo,
    __bf16* __restrict__ qi_hi, __bf16* __restrict__ qi_lo,
    __bf16* __restrict__ kr_hi, __bf16* __restrict__ kr_lo,
    __bf16* __restrict__ ki_hi, __bf16* __restrict__ ki_lo,
    __bf16* __restrict__ vr_hi, __bf16* __restrict__ vr_lo,
    __bf16* __restrict__ vi_hi, __bf16* __restrict__ vi_lo) {
  const int wave = threadIdx.x >> 5;
  const int lane = threadIdx.x & 31;
  const int hlf  = lane >> 4;
  const int col  = lane & 15;
  const int n0   = blockIdx.x * 32;
  const int m0   = (blockIdx.y * 8 + wave) * 32;
  const int proj = blockIdx.z;   // 0=Q, 1=K, 2=V

  const __bf16* wrh = (proj == 0) ? wqr_hi : (proj == 1) ? wkr_hi : wvr_hi;
  const __bf16* wrl = (proj == 0) ? wqr_lo : (proj == 1) ? wkr_lo : wvr_lo;
  const __bf16* wih = (proj == 0) ? wqi_hi : (proj == 1) ? wki_hi : wvi_hi;
  const __bf16* wil = (proj == 0) ? wqi_lo : (proj == 1) ? wki_lo : wvi_lo;

  v8f accP[2][2], accQ[2][2], accR[2][2], accT[2][2];
#pragma unroll
  for (int mi = 0; mi < 2; ++mi)
#pragma unroll
    for (int ni = 0; ni < 2; ++ni) {
      accP[mi][ni] = vzero8(); accQ[mi][ni] = vzero8();
      accR[mi][ni] = vzero8(); accT[mi][ni] = vzero8();
    }

  for (int kc = 0; kc < D_MODEL / 32; ++kc) {
    const int k0 = kc * 32;
    Frag axr[2], axi[2], bwr[2], bwi[2];
#pragma unroll
    for (int mi = 0; mi < 2; ++mi) {
      axr[mi] = load_a_frag(xr_hi, xr_lo, D_MODEL, m0 + 16 * mi, k0);
      axi[mi] = load_a_frag(xi_hi, xi_lo, D_MODEL, m0 + 16 * mi, k0);
    }
#pragma unroll
    for (int ni = 0; ni < 2; ++ni) {
      bwr[ni] = load_b_frag(wrh, wrl, D_MODEL, n0 + 16 * ni, k0);
      bwi[ni] = load_b_frag(wih, wil, D_MODEL, n0 + 16 * ni, k0);
    }
#pragma unroll
    for (int mi = 0; mi < 2; ++mi)
#pragma unroll
      for (int ni = 0; ni < 2; ++ni) {
        accP[mi][ni] = wmma_bf16x3(axr[mi], bwr[ni], accP[mi][ni]);  // xr*wr
        accQ[mi][ni] = wmma_bf16x3(axi[mi], bwi[ni], accQ[mi][ni]);  // xi*wi
        accR[mi][ni] = wmma_bf16x3(axr[mi], bwi[ni], accR[mi][ni]);  // xr*wi
        accT[mi][ni] = wmma_bf16x3(axi[mi], bwr[ni], accT[mi][ni]);  // xi*wr
      }
  }

#pragma unroll
  for (int mi = 0; mi < 2; ++mi)
#pragma unroll
    for (int ni = 0; ni < 2; ++ni)
#pragma unroll
      for (int i = 0; i < 8; ++i) {
        float yre = accP[mi][ni][i] - accQ[mi][ni][i];
        float yim = accR[mi][ni][i] + accT[mi][ni][i];
        const int m  = m0 + 16 * mi + i + 8 * hlf;  // C layout: row = VGPR + 8*half
        const int n  = n0 + 16 * ni + col;          //           col = lane%16
        const int b  = m >> 10, s = m & (SEQ - 1);
        const int h  = n >> 6,  dh = n & (DH - 1);
        const size_t bh = (size_t)b * NHEADS + h;
        if (proj == 2) {
          const size_t idx = (bh * DH + dh) * SEQ + s;   // V transposed [B,H,Dh,S]
          store_split(vr_hi, vr_lo, idx, yre);
          store_split(vi_hi, vi_lo, idx, yim);
        } else {
          // rotor e^{i * s * 10000^{-dh/Dh}}
          const float ang = (float)s * __expf(-(float)dh * 0.14391565033f); // ln(1e4)/64
          float sn, cs; __sincosf(ang, &sn, &cs);
          const float rr = yre * cs - yim * sn;
          const float ri = yre * sn + yim * cs;
          const size_t idx = (bh * SEQ + s) * DH + dh;   // [B,H,S,Dh]
          if (proj == 0) { store_split(qr_hi, qr_lo, idx, rr); store_split(qi_hi, qi_lo, idx, ri); }
          else           { store_split(kr_hi, kr_lo, idx, rr); store_split(ki_hi, ki_lo, idx, ri); }
        }
      }
}

// ---------------------------------------------------------------------------
// Kernel 2: flash-style complex attention per (b,h).
// Per wave: 16 query rows; stream kv in chunks of 32 (= WMMA K).
//   scores: re = Qr*Kr + Qi*Ki ; im = Qi*Kr - Qr*Ki   (Q * conj(K), scaled)
//   online softmax of re; weights w = p * (cos(im), sin(im))
//   O_re += Wre*Vr - Wim*Vi ; O_im += Wre*Vi + Wim*Vr
// The data-dependent W goes C-layout -> LDS -> A-layout (lane transpose; LDS
// ops are in-order within a wave).  Im(W) is pre-negated into its own
// fragment since bf16 WMMA only supports C-negation (ISA NEG = CNeg,0,0).
// Context written as bf16 hi/lo planes for the output projection.
// ---------------------------------------------------------------------------
__global__ __launch_bounds__(128) void attn_kernel(
    const __bf16* __restrict__ qr_hi, const __bf16* __restrict__ qr_lo,
    const __bf16* __restrict__ qi_hi, const __bf16* __restrict__ qi_lo,
    const __bf16* __restrict__ kr_hi, const __bf16* __restrict__ kr_lo,
    const __bf16* __restrict__ ki_hi, const __bf16* __restrict__ ki_lo,
    const __bf16* __restrict__ vr_hi, const __bf16* __restrict__ vr_lo,
    const __bf16* __restrict__ vi_hi, const __bf16* __restrict__ vi_lo,
    __bf16* __restrict__ ar_hi, __bf16* __restrict__ ar_lo,
    __bf16* __restrict__ ai_hi, __bf16* __restrict__ ai_lo) {
  const int wave = threadIdx.x >> 5;
  const int lane = threadIdx.x & 31;
  const int hlf  = lane >> 4;
  const int col  = lane & 15;
  const int bh   = blockIdx.z;
  const int b    = bh >> 4, h = bh & 15;
  const int s0   = (blockIdx.x * 4 + wave) * 16;
  const float scale = 0.125f;   // 1/sqrt(Dh)

  const size_t qkoff = (size_t)bh * SEQ * DH;   // same extent for V^T planes
  const __bf16* Qrh = qr_hi + qkoff; const __bf16* Qrl = qr_lo + qkoff;
  const __bf16* Qih = qi_hi + qkoff; const __bf16* Qil = qi_lo + qkoff;
  const __bf16* Krh = kr_hi + qkoff; const __bf16* Krl = kr_lo + qkoff;
  const __bf16* Kih = ki_hi + qkoff; const __bf16* Kil = ki_lo + qkoff;
  const __bf16* Vrh = vr_hi + qkoff; const __bf16* Vrl = vr_lo + qkoff;
  const __bf16* Vih = vi_hi + qkoff; const __bf16* Vil = vi_lo + qkoff;

  __shared__ float wlds[4][2][16 * 32];   // per-wave 16x32 complex W staging
  float* wre_l = wlds[wave][0];
  float* wim_l = wlds[wave][1];

  // persistent Q fragments: 2 K-chunks x {re,im}
  Frag fqr[2], fqi[2];
#pragma unroll
  for (int kc = 0; kc < 2; ++kc) {
    fqr[kc] = load_a_frag(Qrh, Qrl, DH, s0, kc * 32);
    fqi[kc] = load_a_frag(Qih, Qil, DH, s0, kc * 32);
  }

  v8f o_re[4], o_im[4];
#pragma unroll
  for (int d = 0; d < 4; ++d) { o_re[d] = vzero8(); o_im[d] = vzero8(); }
  float mrow[8], lrow[8];
#pragma unroll
  for (int i = 0; i < 8; ++i) { mrow[i] = -1e30f; lrow[i] = 0.0f; }

  for (int tc = 0; tc < SEQ / 32; ++tc) {
    const int t0 = tc * 32;

    // ---- scores for two 16x16 subtiles (t0..t0+15, t0+16..t0+31) ----
    v8f sre[2], sim[2];
#pragma unroll
    for (int st = 0; st < 2; ++st) {
      v8f P = vzero8(), Qc = vzero8(), R = vzero8(), T = vzero8();
#pragma unroll
      for (int kc = 0; kc < 2; ++kc) {
        Frag bkr = load_b_frag(Krh, Krl, DH, t0 + st * 16, kc * 32);
        Frag bki = load_b_frag(Kih, Kil, DH, t0 + st * 16, kc * 32);
        P  = wmma_bf16x3(fqr[kc], bkr, P);    // Qr*Kr
        Qc = wmma_bf16x3(fqi[kc], bki, Qc);   // Qi*Ki
        R  = wmma_bf16x3(fqi[kc], bkr, R);    // Qi*Kr
        T  = wmma_bf16x3(fqr[kc], bki, T);    // Qr*Ki
      }
#pragma unroll
      for (int i = 0; i < 8; ++i) {
        sre[st][i] = (P[i] + Qc[i]) * scale;
        sim[st][i] = (R[i] - T[i]) * scale;
      }
    }

    // ---- online softmax: row max/alpha (row = i + 8*half, 16 lanes = cols) ----
    float alpha[8], mnew[8];
#pragma unroll
    for (int i = 0; i < 8; ++i) {
      float v = fmaxf(sre[0][i], sre[1][i]);
#pragma unroll
      for (int off = 1; off < 16; off <<= 1)
        v = fmaxf(v, __shfl_xor(v, off, 16));
      mnew[i]  = fmaxf(mrow[i], v);
      alpha[i] = __expf(mrow[i] - mnew[i]);
      mrow[i]  = mnew[i];
    }

    // ---- probs * phase -> LDS (C layout), row sums ----
#pragma unroll
    for (int i = 0; i < 8; ++i) {
      float psum = 0.0f;
      const int row = i + 8 * hlf;
#pragma unroll
      for (int st = 0; st < 2; ++st) {
        const float p = __expf(sre[st][i] - mnew[i]);
        float sn, cs; __sincosf(sim[st][i], &sn, &cs);
        wre_l[row * 32 + st * 16 + col] = p * cs;
        wim_l[row * 32 + st * 16 + col] = p * sn;
        psum += p;
      }
#pragma unroll
      for (int off = 1; off < 16; off <<= 1)
        psum += __shfl_xor(psum, off, 16);
      lrow[i] = lrow[i] * alpha[i] + psum;
    }

    // ---- lane-transpose: read W back in A-fragment layout ----
    Frag fwr, fwi, fwin;
    {
      const float* pr = wre_l + (size_t)col * 32;  // A row = lane%16
      const float* pi = wim_l + (size_t)col * 32;
#pragma unroll
      for (int e = 0; e < 8; ++e) {
        const int k = 8 * hlf + e;
        frag_set(fwr, e, pr[k]);
        const float wiv = pi[k];
        frag_set(fwi, e, wiv);
        frag_set(fwin, e, -wiv);
      }
#pragma unroll
      for (int e = 0; e < 8; ++e) {
        const int k = 16 + 8 * hlf + e;
        frag_set(fwr, 8 + e, pr[k]);
        const float wiv = pi[k];
        frag_set(fwi, 8 + e, wiv);
        frag_set(fwin, 8 + e, -wiv);
      }
    }

    // ---- O rescale + W x V accumulation over 4 dh-chunks ----
#pragma unroll
    for (int d = 0; d < 4; ++d) {
      Frag bvr = load_b_frag(Vrh, Vrl, SEQ, d * 16, t0);  // V^T [Dh,S]
      Frag bvi = load_b_frag(Vih, Vil, SEQ, d * 16, t0);
#pragma unroll
      for (int i = 0; i < 8; ++i) { o_re[d][i] *= alpha[i]; o_im[d][i] *= alpha[i]; }
      o_re[d] = wmma_bf16x3(fwr,  bvr, o_re[d]);   // +Wre*Vr
      o_re[d] = wmma_bf16x3(fwin, bvi, o_re[d]);   // -Wim*Vi
      o_im[d] = wmma_bf16x3(fwr,  bvi, o_im[d]);   // +Wre*Vi
      o_im[d] = wmma_bf16x3(fwi,  bvr, o_im[d]);   // +Wim*Vr
    }
  }

  // ---- normalize by l, write context [B,S,D] as bf16 hi/lo planes ----
#pragma unroll
  for (int d = 0; d < 4; ++d) {
#pragma unroll
    for (int i = 0; i < 8; ++i) {
      const float inv = 1.0f / lrow[i];
      const int s = s0 + i + 8 * hlf;
      const int n = h * DH + d * 16 + col;
      const size_t idx = ((size_t)(b * SEQ + s)) * D_MODEL + n;
      store_split(ar_hi, ar_lo, idx, o_re[d][i] * inv);
      store_split(ai_hi, ai_lo, idx, o_im[d][i] * inv);
    }
  }
}

// ---------------------------------------------------------------------------
// Kernel 3: complex output projection, 2x2 register blocked, writes
// interleaved [B,S,D,2] fp32.
// ---------------------------------------------------------------------------
__global__ __launch_bounds__(256) void out_proj_kernel(
    const __bf16* __restrict__ ar_hi, const __bf16* __restrict__ ar_lo,
    const __bf16* __restrict__ ai_hi, const __bf16* __restrict__ ai_lo,
    const __bf16* __restrict__ wor_hi, const __bf16* __restrict__ wor_lo,
    const __bf16* __restrict__ woi_hi, const __bf16* __restrict__ woi_lo,
    float* __restrict__ out) {
  const int wave = threadIdx.x >> 5;
  const int lane = threadIdx.x & 31;
  const int hlf  = lane >> 4;
  const int col  = lane & 15;
  const int n0   = blockIdx.x * 32;
  const int m0   = (blockIdx.y * 8 + wave) * 32;

  v8f accP[2][2], accQ[2][2], accR[2][2], accT[2][2];
#pragma unroll
  for (int mi = 0; mi < 2; ++mi)
#pragma unroll
    for (int ni = 0; ni < 2; ++ni) {
      accP[mi][ni] = vzero8(); accQ[mi][ni] = vzero8();
      accR[mi][ni] = vzero8(); accT[mi][ni] = vzero8();
    }

  for (int kc = 0; kc < D_MODEL / 32; ++kc) {
    const int k0 = kc * 32;
    Frag aar[2], aai[2], bwr[2], bwi[2];
#pragma unroll
    for (int mi = 0; mi < 2; ++mi) {
      aar[mi] = load_a_frag(ar_hi, ar_lo, D_MODEL, m0 + 16 * mi, k0);
      aai[mi] = load_a_frag(ai_hi, ai_lo, D_MODEL, m0 + 16 * mi, k0);
    }
#pragma unroll
    for (int ni = 0; ni < 2; ++ni) {
      bwr[ni] = load_b_frag(wor_hi, wor_lo, D_MODEL, n0 + 16 * ni, k0);
      bwi[ni] = load_b_frag(woi_hi, woi_lo, D_MODEL, n0 + 16 * ni, k0);
    }
#pragma unroll
    for (int mi = 0; mi < 2; ++mi)
#pragma unroll
      for (int ni = 0; ni < 2; ++ni) {
        accP[mi][ni] = wmma_bf16x3(aar[mi], bwr[ni], accP[mi][ni]);
        accQ[mi][ni] = wmma_bf16x3(aai[mi], bwi[ni], accQ[mi][ni]);
        accR[mi][ni] = wmma_bf16x3(aar[mi], bwi[ni], accR[mi][ni]);
        accT[mi][ni] = wmma_bf16x3(aai[mi], bwr[ni], accT[mi][ni]);
      }
  }

#pragma unroll
  for (int mi = 0; mi < 2; ++mi)
#pragma unroll
    for (int ni = 0; ni < 2; ++ni)
#pragma unroll
      for (int i = 0; i < 8; ++i) {
        const int m = m0 + 16 * mi + i + 8 * hlf;
        const int n = n0 + 16 * ni + col;
        const size_t idx = ((size_t)m * D_MODEL + n) * 2;
        out[idx]     = accP[mi][ni][i] - accQ[mi][ni][i];   // real
        out[idx + 1] = accR[mi][ni][i] + accT[mi][ni][i];   // imag
      }
}

// ---------------------------------------------------------------------------
// Host launcher.  Workspace: bf16 hi/lo planes.
//   20 planes of XN = 2048*1024 elements (x, Q, K, V^T, context)
//   16 planes of WN = 1024*1024 elements (wq, wk, wv, wo  x re/im x hi/lo)
// Total ~112 MB.
// ---------------------------------------------------------------------------
extern "C" void kernel_launch(void* const* d_in, const int* in_sizes, int n_in,
                              void* d_out, int out_size, void* d_ws, size_t ws_size,
                              hipStream_t stream) {
  (void)in_sizes; (void)n_in; (void)out_size; (void)ws_size;
  const float* fin[10];
  for (int i = 0; i < 10; ++i) fin[i] = (const float*)d_in[i];
  // order: x_re x_im wq_re wq_im wk_re wk_im wv_re wv_im wo_re wo_im

  const size_t XN = (size_t)MROWS * D_MODEL;   // 2,097,152
  const size_t WN = (size_t)D_MODEL * D_MODEL; // 1,048,576

  __bf16* p = (__bf16*)d_ws;
  // x planes
  __bf16* xr_hi = p; p += XN;  __bf16* xr_lo = p; p += XN;
  __bf16* xi_hi = p; p += XN;  __bf16* xi_lo = p; p += XN;
  // weight planes: [wq,wk,wv,wo] x [re_hi, re_lo, im_hi, im_lo]
  __bf16* wpl[16];
  for (int i = 0; i < 16; ++i) { wpl[i] = p; p += WN; }
  // Q/K/V^T planes
  __bf16* qr_hi = p; p += XN;  __bf16* qr_lo = p; p += XN;
  __bf16* qi_hi = p; p += XN;  __bf16* qi_lo = p; p += XN;
  __bf16* kr_hi = p; p += XN;  __bf16* kr_lo = p; p += XN;
  __bf16* ki_hi = p; p += XN;  __bf16* ki_lo = p; p += XN;
  __bf16* vr_hi = p; p += XN;  __bf16* vr_lo = p; p += XN;
  __bf16* vi_hi = p; p += XN;  __bf16* vi_lo = p; p += XN;
  // context planes
  __bf16* ar_hi = p; p += XN;  __bf16* ar_lo = p; p += XN;
  __bf16* ai_hi = p; p += XN;  __bf16* ai_lo = p; p += XN;

  // 0) pre-split shared operands into bf16 hi/lo planes (memory bound)
  split_bf16_kernel<<<dim3((unsigned)(XN / 4 / 256)), 256, 0, stream>>>(fin[0], xr_hi, xr_lo, (int)(XN / 4));
  split_bf16_kernel<<<dim3((unsigned)(XN / 4 / 256)), 256, 0, stream>>>(fin[1], xi_hi, xi_lo, (int)(XN / 4));
  for (int w = 0; w < 4; ++w) {   // wq, wk, wv, wo
    split_bf16_kernel<<<dim3((unsigned)(WN / 4 / 256)), 256, 0, stream>>>(
        fin[2 + 2 * w], wpl[4 * w + 0], wpl[4 * w + 1], (int)(WN / 4));      // re
    split_bf16_kernel<<<dim3((unsigned)(WN / 4 / 256)), 256, 0, stream>>>(
        fin[3 + 2 * w], wpl[4 * w + 2], wpl[4 * w + 3], (int)(WN / 4));      // im
  }

  // 1) QKV projection + RoPE: grid = (N tiles/32, M tiles/256, {q,k,v})
  qkv_rope_kernel<<<dim3(D_MODEL / 32, MROWS / 256, 3), 256, 0, stream>>>(
      xr_hi, xr_lo, xi_hi, xi_lo,
      wpl[0], wpl[1], wpl[2], wpl[3],      // wq re/im hi/lo
      wpl[4], wpl[5], wpl[6], wpl[7],      // wk
      wpl[8], wpl[9], wpl[10], wpl[11],    // wv
      qr_hi, qr_lo, qi_hi, qi_lo,
      kr_hi, kr_lo, ki_hi, ki_lo,
      vr_hi, vr_lo, vi_hi, vi_lo);

  // 2) Attention: grid = (query tile groups, 1, B*H); 4 waves/block
  attn_kernel<<<dim3(SEQ / 64, 1, BATCH * NHEADS), 128, 0, stream>>>(
      qr_hi, qr_lo, qi_hi, qi_lo,
      kr_hi, kr_lo, ki_hi, ki_lo,
      vr_hi, vr_lo, vi_hi, vi_lo,
      ar_hi, ar_lo, ai_hi, ai_lo);

  // 3) Output projection -> [B,S,D,2]
  out_proj_kernel<<<dim3(D_MODEL / 32, MROWS / 256, 1), 256, 0, stream>>>(
      ar_hi, ar_lo, ai_hi, ai_lo,
      wpl[12], wpl[13], wpl[14], wpl[15],  // wo
      (float*)d_out);
}
